// Network_discrete_10496900072253
// MI455X (gfx1250) — matrix operand order repaired and checked by reading, not compiled, not run on hardware
//
#include <hip/hip_runtime.h>
#include <hip/hip_bf16.h>

// ---------------------------------------------------------------------------
// MI455X (gfx1250) implementation.
//   * All dense layers -> bf16 V_WMMA_F32_16X16X32_BF16, f32 accumulate.
//   * SpMM (roofline-dominant, ~0.9GB/pass) -> f32 atomic scatter into L2-
//     resident accumulators (25.6MB fits 192MB L2).
//   * Weights pre-transposed to bf16 (N-major) so B-fragments are contiguous
//     b128 loads; A-fragments read once from global f32, cvt in-register.
//   * All tile boundaries are multiples of 16 -> waves are fully in-range or
//     exit wave-uniformly; no per-lane guards, EXEC stays all-ones (WMMA req).
// ---------------------------------------------------------------------------

typedef __bf16 bf16_t;
typedef __attribute__((ext_vector_type(16))) __bf16 v16bf;
typedef __attribute__((ext_vector_type(8)))  float  v8f;

#define NN      100000
#define C0      40000
#define C1      30000
#define HID     64
#define KEXP    8
#define DIN     512
#define NCLS    16

__device__ __forceinline__ bf16_t f2bf(float f) {
  unsigned int u = __builtin_bit_cast(unsigned int, f);
  unsigned int r = u + 0x7FFFu + ((u >> 16) & 1u);   // round-to-nearest-even
  unsigned short h = (unsigned short)(r >> 16);
  return __builtin_bit_cast(bf16_t, h);
}

__device__ __forceinline__ v8f wmma_bf16(v16bf a, v16bf b, v8f c) {
  return __builtin_amdgcn_wmma_f32_16x16x32_bf16(false, a, false, b,
                                                 (short)0, c, false, false);
}

// A-fragment (16x32 bf16) from row-major f32; optional 1/max(deg,1) scale.
// ISA layout: lane l: row M = l&15; kb = (l>>4)*8; VGPRs 0-3 hold K=kb..kb+7,
// VGPRs 4-7 hold K=16+kb..16+kb+7 (contiguous -> 2x 32B vector loads).
__device__ __forceinline__ v16bf load_a(const float* A, int lda, int row0,
                                        int kk, const float* deg) {
  int l  = threadIdx.x & 31;
  int m  = row0 + (l & 15);
  int kb = (l >> 4) * 8;
  float s = 1.0f;
  if (deg) s = 1.0f / fmaxf(deg[m], 1.0f);
  const float* p0 = A + (size_t)m * lda + kk + kb;
  const float* p1 = p0 + 16;
  v16bf a;
#pragma unroll
  for (int i = 0; i < 8; ++i) a[i]     = f2bf(p0[i] * s);
#pragma unroll
  for (int i = 0; i < 8; ++i) a[8 + i] = f2bf(p1[i] * s);
  return a;
}

// B-fragment (32x16 bf16) from N-major bf16 weights Bt[n][K].
// lane l: col = n0 + (l&15); kb = (l>>4)*8; K = kk+kb..+7 and kk+16+kb..+7.
__device__ __forceinline__ v16bf load_b(const bf16_t* Bt, int ldb, int n0, int kk) {
  int l = threadIdx.x & 31;
  const bf16_t* p = Bt + (size_t)(n0 + (l & 15)) * ldb + kk + ((l >> 4) * 8);
  v16bf b;
#pragma unroll
  for (int i = 0; i < 8; ++i) { b[i] = p[i]; b[8 + i] = p[16 + i]; }
  return b;
}

// C (16x16 f32): lane l: col = n0 + (l&15); VGPR r holds row r + 8*(l>=16).
__device__ __forceinline__ void store_c(float* C, int ldc, int row0,
                                        int n0, const v8f c) {
  int l    = threadIdx.x & 31;
  int col  = n0 + (l & 15);
  int base = row0 + ((l >> 4) * 8);
#pragma unroll
  for (int r = 0; r < 8; ++r) C[(size_t)(base + r) * ldc + col] = c[r];
}

// ---------------- weight prep: transpose + f32->bf16 ----------------------
__global__ void k_prep(const float* Wpre, const float* opW, const float* fcW,
                       const float* gnnW, const float* outW,
                       bf16_t* WpreT, bf16_t* opWT, bf16_t* fcWT,
                       bf16_t* gnnWT, bf16_t* outWT) {
  int t = blockIdx.x * blockDim.x + threadIdx.x;
  if (t < HID * DIN) { int n = t / DIN, k = t % DIN; WpreT[t] = f2bf(Wpre[k * HID + n]); }
  if (t < KEXP * HID * HID) {
    int g = t / (HID * HID), r = t % (HID * HID), n = r / HID, k = r % HID;
    opWT[t] = f2bf(opW[g * HID * HID + k * HID + n]);
  }
  if (t < 2 * HID * HID) {
    int g = t / (HID * HID), r = t % (HID * HID), n = r / HID, k = r % HID;
    fcWT[t] = f2bf(fcW[g * HID * HID + k * HID + n]);
  }
  if (t < HID * HID) { int n = t / HID, k = t % HID; gnnWT[t] = f2bf(gnnW[k * HID + n]); }
  if (t < NCLS * HID) { int n = t / HID, k = t % HID; outWT[t] = f2bf(outW[k * HID + n]); }
}

// ---------------- pre layer: h0[0:C0] = feats0 @ W_pre + b ---------------
__global__ void k_gemm_pre(const float* __restrict__ A, const bf16_t* __restrict__ Bt,
                           const float* __restrict__ bias, float* __restrict__ C) {
  int w    = threadIdx.x >> 5;
  int row0 = blockIdx.x * 128 + w * 16;
  if (row0 >= C0) return;                       // wave-uniform exit
  int l = threadIdx.x & 31;
  v8f c[4];
#pragma unroll
  for (int t = 0; t < 4; ++t) {
    float bv = bias[t * 16 + (l & 15)];
#pragma unroll
    for (int r = 0; r < 8; ++r) c[t][r] = bv;
  }
  for (int kk = 0; kk < DIN; kk += 32) {
    v16bf a = load_a(A, DIN, row0, kk, nullptr);
#pragma unroll
    for (int t = 0; t < 4; ++t) c[t] = wmma_bf16(a, load_b(Bt, DIN, t * 16, kk), c[t]);
  }
#pragma unroll
  for (int t = 0; t < 4; ++t) store_c(C, HID, row0, t * 16, c[t]);
}

// ---------------- edge-list SpMM scatter (roofline stage) -----------------
__global__ void k_spmm(const int* __restrict__ src, const int* __restrict__ dst,
                       const float* __restrict__ h, float* __restrict__ agg,
                       float* __restrict__ deg, int E, int do_deg) {
  long long t = (long long)blockIdx.x * blockDim.x + threadIdx.x;
  long long e = t >> 4;
  if (e >= E) return;
  int c4 = (int)(t & 15) * 4;
  int s = src[e], d = dst[e];
  const float4 v = *(const float4*)(h + (size_t)s * HID + c4);
  float* p = agg + (size_t)d * HID + c4;
  atomicAdd(p + 0, v.x); atomicAdd(p + 1, v.y);
  atomicAdd(p + 2, v.z); atomicAdd(p + 3, v.w);
  if (do_deg && c4 == 0) atomicAdd(deg + d, 1.0f);
}

// ------- grouped op_W stage: all 8 experts via WMMA + per-row select ------
__global__ void k_gcn(const float* __restrict__ agg, const float* __restrict__ deg,
                      const bf16_t* __restrict__ opWT, const float* __restrict__ op_b,
                      const float* __restrict__ h0, const float* __restrict__ embW,
                      const float* __restrict__ embB, const int* __restrict__ assign,
                      float* __restrict__ h_att, float* __restrict__ h_t) {
  int w    = threadIdx.x >> 5;
  int row0 = blockIdx.x * 128 + w * 16;
  if (row0 >= NN) return;                       // wave-uniform exit
  int l    = threadIdx.x & 31;
  int rb   = row0 + ((l >> 4) * 8);             // first of this lane's 8 rows

  v16bf a0 = load_a(agg, HID, row0, 0,  deg);
  v16bf a1 = load_a(agg, HID, row0, 32, deg);

  int as[8];
#pragma unroll
  for (int r = 0; r < 8; ++r) as[r] = assign[rb + r];

  v8f res[4];
#pragma unroll
  for (int t = 0; t < 4; ++t)
#pragma unroll
    for (int r = 0; r < 8; ++r) res[t][r] = 0.0f;

#pragma unroll
  for (int k = 0; k < KEXP; ++k) {
    const bf16_t* Bt = opWT + (size_t)k * HID * HID;
#pragma unroll
    for (int t = 0; t < 4; ++t) {
      float bv = op_b[k * HID + t * 16 + (l & 15)];
      v8f c;
#pragma unroll
      for (int r = 0; r < 8; ++r) c[r] = bv;
      c = wmma_bf16(a0, load_b(Bt, HID, t * 16, 0),  c);
      c = wmma_bf16(a1, load_b(Bt, HID, t * 16, 32), c);
#pragma unroll
      for (int r = 0; r < 8; ++r) res[t][r] += (as[r] == k) ? c[r] : 0.0f;
    }
  }

#pragma unroll
  for (int t = 0; t < 4; ++t) {
    int col = t * 16 + (l & 15);
#pragma unroll
    for (int r = 0; r < 8; ++r) {
      int m = rb + r;
      float v = res[t][r] + h0[(size_t)m * HID + col];
      if (m >= C0) {
        int seg = (m < C0 + C1) ? 0 : 1;
        int idx = m - (seg ? (C0 + C1) : C0);
        v += embW[((size_t)seg * C1 + idx) * HID + col] + embB[seg * HID + col];
      }
      h_att[(size_t)m * HID + col] = v;
      if (m < C0) h_t[(size_t)m * HID + col] = v;   // pass-through segment
    }
  }
}

// ---------------- fc layers on the two embedding segments -----------------
__global__ void k_fc(const float* __restrict__ h_att, const bf16_t* __restrict__ fcWT,
                     const float* __restrict__ fc_b, float* __restrict__ h_t) {
  int seg  = blockIdx.y;
  int base = C0 + seg * C1;
  int w    = threadIdx.x >> 5;
  int row0 = base + blockIdx.x * 128 + w * 16;
  if (row0 >= base + C1) return;                // wave-uniform exit
  int l = threadIdx.x & 31;
  const bf16_t* Bt = fcWT + (size_t)seg * HID * HID;
  v8f c[4];
#pragma unroll
  for (int t = 0; t < 4; ++t) {
    float bv = fc_b[seg * HID + t * 16 + (l & 15)];
#pragma unroll
    for (int r = 0; r < 8; ++r) c[t][r] = bv;
  }
#pragma unroll
  for (int kk = 0; kk < HID; kk += 32) {
    v16bf a = load_a(h_att, HID, row0, kk, nullptr);
#pragma unroll
    for (int t = 0; t < 4; ++t) c[t] = wmma_bf16(a, load_b(Bt, HID, t * 16, kk), c[t]);
  }
#pragma unroll
  for (int t = 0; t < 4; ++t) store_c(h_t, HID, row0, t * 16, c[t]);
}

// -------- fused tail: gnn GEMM + ELU -> node_emb; out GEMM -> logits ------
__global__ void k_final(const float* __restrict__ agg2, const float* __restrict__ deg,
                        const bf16_t* __restrict__ gnnWT, const float* __restrict__ gnn_b,
                        const bf16_t* __restrict__ outWT, const float* __restrict__ out_b,
                        float* __restrict__ out) {
  int w    = threadIdx.x >> 5;
  int row0 = blockIdx.x * 128 + w * 16;
  if (row0 >= NN) return;                       // wave-uniform exit
  int l  = threadIdx.x & 31;
  int rb = row0 + ((l >> 4) * 8);

  v16bf a0 = load_a(agg2, HID, row0, 0,  deg);
  v16bf a1 = load_a(agg2, HID, row0, 32, deg);

  float* emb = out;                             // node_emb: N x 64
#pragma unroll
  for (int t = 0; t < 4; ++t) {
    int col = t * 16 + (l & 15);
    float bv = gnn_b[col];
    v8f c;
#pragma unroll
    for (int r = 0; r < 8; ++r) c[r] = bv;
    c = wmma_bf16(a0, load_b(gnnWT, HID, t * 16, 0),  c);
    c = wmma_bf16(a1, load_b(gnnWT, HID, t * 16, 32), c);
#pragma unroll
    for (int r = 0; r < 8; ++r) {
      float x = c[r];
      x = (x > 0.0f) ? x : (__expf(x) - 1.0f);  // ELU
      emb[(size_t)(rb + r) * HID + col] = x;
    }
  }

  // Same-wave store->load on own 16x64 tile: force stores complete, then
  // re-pack as A-fragment for the logits GEMM (ISA: same-wave store->load
  // to same address kept in order; storecnt wait makes it explicit).
  asm volatile("s_wait_storecnt 0x0" ::: "memory");

  v16bf e0 = load_a(emb, HID, row0, 0,  nullptr);
  v16bf e1 = load_a(emb, HID, row0, 32, nullptr);

  float bv = out_b[l & 15];
  v8f c2;
#pragma unroll
  for (int r = 0; r < 8; ++r) c2[r] = bv;
  c2 = wmma_bf16(e0, load_b(outWT, HID, 0, 0),  c2);
  c2 = wmma_bf16(e1, load_b(outWT, HID, 0, 32), c2);

  float* lg1 = out + (size_t)NN * HID;
  float* lg2 = lg1 + (size_t)NN * NCLS;
  int col = l & 15;
#pragma unroll
  for (int r = 0; r < 8; ++r) {
    float v = c2[r];
    lg1[(size_t)(rb + r) * NCLS + col] = v;
    lg2[(size_t)(rb + r) * NCLS + col] = v;
  }
}

// ---------------------------------------------------------------------------
extern "C" void kernel_launch(void* const* d_in, const int* in_sizes, int n_in,
                              void* d_out, int out_size, void* d_ws, size_t ws_size,
                              hipStream_t stream) {
  const float* feats0 = (const float*)d_in[0];
  const float* W_pre  = (const float*)d_in[1];
  const float* b_pre  = (const float*)d_in[2];
  const float* emb_W  = (const float*)d_in[3];
  const float* emb_b  = (const float*)d_in[4];
  const float* op_W   = (const float*)d_in[5];
  const float* op_b   = (const float*)d_in[6];
  const float* fc_W   = (const float*)d_in[7];
  const float* fc_b   = (const float*)d_in[8];
  const float* gnn_W  = (const float*)d_in[9];
  const float* gnn_b  = (const float*)d_in[10];
  const float* out_W  = (const float*)d_in[11];
  const float* out_b  = (const float*)d_in[12];
  const int*   src    = (const int*)d_in[13];
  const int*   dst    = (const int*)d_in[14];
  const int*   assign = (const int*)d_in[15];
  const int E = in_sizes[13];

  // Workspace layout (f32 unless noted)
  float* deg   = (float*)d_ws;                 // N
  float* agg   = deg   + NN;                   // N*64
  float* agg2  = agg   + (size_t)NN * HID;     // N*64
  float* h0    = agg2  + (size_t)NN * HID;     // N*64
  float* h_att = h0    + (size_t)NN * HID;     // N*64
  float* h_t   = h_att + (size_t)NN * HID;     // N*64
  bf16_t* WpreT = (bf16_t*)(h_t + (size_t)NN * HID);
  bf16_t* opWT  = WpreT + HID * DIN;
  bf16_t* fcWT  = opWT  + KEXP * HID * HID;
  bf16_t* gnnWT = fcWT  + 2 * HID * HID;
  bf16_t* outWT = gnnWT + HID * HID;

  // Per-replay zeroing: deg + agg + agg2 (atomic accumulators) and the
  // h0 rows not overwritten by the pre-GEMM ([C0, N)).
  hipMemsetAsync(deg, 0, (size_t)NN * (1 + 2 * HID) * sizeof(float), stream);
  hipMemsetAsync(h0 + (size_t)C0 * HID, 0,
                 (size_t)(NN - C0) * HID * sizeof(float), stream);

  k_prep<<<(HID * DIN + 255) / 256, 256, 0, stream>>>(
      W_pre, op_W, fc_W, gnn_W, out_W, WpreT, opWT, fcWT, gnnWT, outWT);

  k_gemm_pre<<<(C0 + 127) / 128, 256, 0, stream>>>(feats0, WpreT, b_pre, h0);

  unsigned gsp = (unsigned)(((long long)E * 16 + 255) / 256);
  k_spmm<<<gsp, 256, 0, stream>>>(src, dst, h0, agg, deg, E, 1);

  k_gcn<<<(NN + 127) / 128, 256, 0, stream>>>(
      agg, deg, opWT, op_b, h0, emb_W, emb_b, assign, h_att, h_t);

  dim3 gfc((C1 + 127) / 128, 2);
  k_fc<<<gfc, 256, 0, stream>>>(h_att, fcWT, fc_b, h_t);

  k_spmm<<<gsp, 256, 0, stream>>>(src, dst, h_t, agg2, deg, E, 0);

  k_final<<<(NN + 127) / 128, 256, 0, stream>>>(
      agg2, deg, gnnWT, gnn_b, outWT, out_b, (float*)d_out);
}
